// Attentive_FFNN_8744553415043
// MI455X (gfx1250) — matrix operand order repaired and checked by reading, not compiled, not run on hardware
//
#include <hip/hip_runtime.h>
#include <hip/hip_bf16.h>
#include <math.h>

#define B_   32
#define S_   512
#define DIN_ 32
#define EMB_ 256
#define H_   8
#define L_   4
#define DFF_ 1024
#define NROW (B_*S_)   // 16384

// ---- gfx1250 async global->LDS copy (guarded; fallback = staged copy) -----
#if defined(__has_builtin)
#if __has_builtin(__builtin_amdgcn_global_load_async_to_lds_b128) && \
    __has_builtin(__builtin_amdgcn_s_wait_asynccnt)
#define USE_ASYNC_COPY 1
#endif
#endif
#ifndef USE_ASYNC_COPY
#define USE_ASYNC_COPY 0
#endif

typedef int v4i_ __attribute__((vector_size(16)));

#if USE_ASYNC_COPY
// builtin signature (from hipcc diagnostic): (v4i*, v4i*, imm int, imm int)
#define ASYNC_B128(gp, lp) \
  __builtin_amdgcn_global_load_async_to_lds_b128((v4i_*)(gp), (v4i_*)(lp), 0, 0)
#define ASYNC_WAIT() __builtin_amdgcn_s_wait_asynccnt(0)
#endif

typedef __attribute__((ext_vector_type(16))) __bf16 v16bf;
typedef __attribute__((ext_vector_type(8)))  __bf16 v8bf;
typedef __attribute__((ext_vector_type(8)))  float  v8f;

union AF { v16bf v; v8bf h[2]; };

__device__ __forceinline__ __bf16 f2bf(float f) {
  unsigned u = __builtin_bit_cast(unsigned, f);
  unsigned r = u + 0x7FFFu + ((u >> 16) & 1u);       // round-to-nearest-even
  unsigned short hh = (unsigned short)(r >> 16);
  return __builtin_bit_cast(__bf16, hh);
}

// ---------------------------------------------------------------------------
// Weight prep: out[l][n][k] = bf16(in[l][k][n])  (convert + transpose)
// ---------------------------------------------------------------------------
__global__ __launch_bounds__(256) void cvt_t_kernel(
    const float* __restrict__ in, __bf16* __restrict__ out, int K, int N)
{
  int o = blockIdx.x * 256 + threadIdx.x;            // over L*K*N (grid-sized)
  int sl = K * N;
  int l  = o / sl;
  int r  = o - l * sl;
  int n  = r / K;
  int kk = r - n * K;
  out[o] = f2bf(in[(size_t)l * sl + (size_t)kk * N + n]);
}

// ---------------------------------------------------------------------------
// Embed: x = src @ Win + b_in + positional-encoding  (f32 + bf16 shadow)
// ---------------------------------------------------------------------------
__global__ __launch_bounds__(256) void embed_kernel(
    const float* __restrict__ src, const float* __restrict__ Win,
    const float* __restrict__ b_in, float* __restrict__ x, __bf16* __restrict__ xb)
{
  int idx = blockIdx.x * 256 + threadIdx.x;          // NROW*EMB threads
  int row = idx >> 8, e = idx & 255;
  const float* sp = src + (size_t)row * DIN_;
  float acc = b_in[e];
#pragma unroll
  for (int kk = 0; kk < DIN_; ++kk) acc += sp[kk] * Win[kk * EMB_ + e];
  float t = sp[0];
  const float C1 = -2.0f * logf(0.0375f) / (float)EMB_;  // constant-folded
  int i = e >> 1;
  float ang = t * __expf((float)i * C1);
  float pe = 0.5f * ((e & 1) ? __cosf(ang) : __sinf(ang));
  float o = acc + pe;
  x[idx]  = o;
  xb[idx] = f2bf(o);
}

// ---------------------------------------------------------------------------
// Double-buffered bf16 WMMA GEMM: C = act(A[M,K] @ W + bias)
// A bf16 [M,K]; Wt bf16 pre-transposed [N,K].
// Block 256 threads = 8 waves (4 row x 2 col); block tile 256x128;
// wave tile 64x64 (4x4 WMMA 16x16x32 tiles); one barrier per k-step.
// Tile fill uses GLOBAL_LOAD_ASYNC_TO_LDS_B128 when available.
// ---------------------------------------------------------------------------
template<bool LEAKY, bool HASBIAS, bool WF32, bool WBF16>
__global__ __launch_bounds__(256) void gemm_wmma(
    const __bf16* __restrict__ A, const __bf16* __restrict__ Wt,
    const float* __restrict__ bias, float* __restrict__ C,
    __bf16* __restrict__ Cb, int M, int N, int K)
{
  __shared__ __bf16 As[2][256][32];   // 32 KB
  __shared__ __bf16 Bs[2][128][32];   // 16 KB

  const int tid  = threadIdx.x;
  const int lane = tid & 31;
  const int wid  = tid >> 5;
  const int wm   = wid & 3, wn = wid >> 2;
  const int half = lane >> 4;
  const int ln   = lane & 15;
  const int rowBase = blockIdx.y * 256;
  const int colBase = blockIdx.x * 128;

  // staging slot coordinates (fixed per thread)
  const int ar0 = tid >> 2, ac0 = (tid & 3) << 3;     // + u*64 rows

  v8f acc[4][4];
#pragma unroll
  for (int mt = 0; mt < 4; ++mt)
#pragma unroll
    for (int nt = 0; nt < 4; ++nt)
#pragma unroll
      for (int e = 0; e < 8; ++e) acc[mt][nt][e] = 0.0f;

#if USE_ASYNC_COPY
  // ---- async DMA tile fill -------------------------------------------------
  {
#pragma unroll
    for (int u = 0; u < 4; ++u)
      ASYNC_B128(&A[(size_t)(rowBase + ar0 + u * 64) * K + ac0], &As[0][ar0 + u * 64][ac0]);
#pragma unroll
    for (int u = 0; u < 2; ++u)
      ASYNC_B128(&Wt[(size_t)(colBase + ar0 + u * 64) * K + ac0], &Bs[0][ar0 + u * 64][ac0]);
  }
  ASYNC_WAIT();
  __syncthreads();
#else
  // ---- staged tile fill ----------------------------------------------------
  v8bf ra[4], rb[2];
#pragma unroll
  for (int u = 0; u < 4; ++u)
    ra[u] = *reinterpret_cast<const v8bf*>(&A[(size_t)(rowBase + ar0 + u * 64) * K + ac0]);
#pragma unroll
  for (int u = 0; u < 2; ++u)
    rb[u] = *reinterpret_cast<const v8bf*>(&Wt[(size_t)(colBase + ar0 + u * 64) * K + ac0]);
#pragma unroll
  for (int u = 0; u < 4; ++u) *reinterpret_cast<v8bf*>(&As[0][ar0 + u * 64][ac0]) = ra[u];
#pragma unroll
  for (int u = 0; u < 2; ++u) *reinterpret_cast<v8bf*>(&Bs[0][ar0 + u * 64][ac0]) = rb[u];
  __syncthreads();
#endif

  const int T = K >> 5;
  for (int t = 0; t < T; ++t) {
    const int buf = t & 1;
#if USE_ASYNC_COPY
    if (t + 1 < T) {
      int k0 = (t + 1) << 5;
      const int nb = 1 - buf;
#pragma unroll
      for (int u = 0; u < 4; ++u)
        ASYNC_B128(&A[(size_t)(rowBase + ar0 + u * 64) * K + k0 + ac0], &As[nb][ar0 + u * 64][ac0]);
#pragma unroll
      for (int u = 0; u < 2; ++u)
        ASYNC_B128(&Wt[(size_t)(colBase + ar0 + u * 64) * K + k0 + ac0], &Bs[nb][ar0 + u * 64][ac0]);
    }
#else
    if (t + 1 < T) {
      int k0 = (t + 1) << 5;
#pragma unroll
      for (int u = 0; u < 4; ++u)
        ra[u] = *reinterpret_cast<const v8bf*>(&A[(size_t)(rowBase + ar0 + u * 64) * K + k0 + ac0]);
#pragma unroll
      for (int u = 0; u < 2; ++u)
        rb[u] = *reinterpret_cast<const v8bf*>(&Wt[(size_t)(colBase + ar0 + u * 64) * K + k0 + ac0]);
    }
#endif

    AF af[4], bf[4];
#pragma unroll
    for (int mt = 0; mt < 4; ++mt) {
      const __bf16* p = &As[buf][wm * 64 + mt * 16 + ln][0];
      af[mt].h[0] = *reinterpret_cast<const v8bf*>(p + half * 8);
      af[mt].h[1] = *reinterpret_cast<const v8bf*>(p + 16 + half * 8);
    }
#pragma unroll
    for (int nt = 0; nt < 4; ++nt) {
      const __bf16* p = &Bs[buf][wn * 64 + nt * 16 + ln][0];
      bf[nt].h[0] = *reinterpret_cast<const v8bf*>(p + half * 8);
      bf[nt].h[1] = *reinterpret_cast<const v8bf*>(p + 16 + half * 8);
    }
#pragma unroll
    for (int mt = 0; mt < 4; ++mt)
#pragma unroll
      for (int nt = 0; nt < 4; ++nt)
        acc[mt][nt] = __builtin_amdgcn_wmma_f32_16x16x32_bf16(
            false, af[mt].v, false, bf[nt].v, (short)0, acc[mt][nt], false, false);

    if (t + 1 < T) {
#if USE_ASYNC_COPY
      ASYNC_WAIT();
#else
      const int nb = 1 - buf;
#pragma unroll
      for (int u = 0; u < 4; ++u) *reinterpret_cast<v8bf*>(&As[nb][ar0 + u * 64][ac0]) = ra[u];
#pragma unroll
      for (int u = 0; u < 2; ++u) *reinterpret_cast<v8bf*>(&Bs[nb][ar0 + u * 64][ac0]) = rb[u];
#endif
      __syncthreads();
    }
  }

#pragma unroll
  for (int nt = 0; nt < 4; ++nt) {
    int col = colBase + wn * 64 + nt * 16 + ln;
    float bv = HASBIAS ? bias[col] : 0.0f;
#pragma unroll
    for (int mt = 0; mt < 4; ++mt) {
#pragma unroll
      for (int e = 0; e < 8; ++e) {
        int row = rowBase + wm * 64 + mt * 16 + e + 8 * half;
        float val = acc[mt][nt][e] + bv;
        if (LEAKY) val = val > 0.0f ? val : 0.01f * val;
        if (WF32)  C [(size_t)row * N + col] = val;
        if (WBF16) Cb[(size_t)row * N + col] = f2bf(val);
      }
    }
  }
}

// ---------------------------------------------------------------------------
// Flash attention with fused residual:
//   x += softmax(QK^T * scale) @ V + RB @ V     (bias added AFTER softmax)
// One block per (b, h, 128 query rows); each wave owns 16 query rows,
// keys streamed in chunks of 64. Q/K/V bf16; x/xb updated in place.
// ---------------------------------------------------------------------------
__global__ __launch_bounds__(256) void attn_kernel(
    const __bf16* __restrict__ q, const __bf16* __restrict__ k,
    const __bf16* __restrict__ v, const float* __restrict__ btab,
    float* __restrict__ x, __bf16* __restrict__ xb)
{
  __shared__ __bf16 Ks[64][32];        // key rows            (4 KB)
  __shared__ __bf16 VsT[32][64];       // V transposed [n][k] (4 KB)
  __shared__ __bf16 Ps[8][16][64];     // per-wave P staging  (16 KB)
  __shared__ float  bt[1024];          // bias column, head h (4 KB)

  const int tid  = threadIdx.x;
  const int lane = tid & 31, wid = tid >> 5;
  const int half = lane >> 4, ln = lane & 15;
  const int bq = blockIdx.x & 3;            // S/128 = 4
  const int h  = (blockIdx.x >> 2) & 7;
  const int b  = blockIdx.x >> 5;
  const int qbase = bq * 128 + wid * 16;

  for (int i = tid; i < 2 * S_ - 1; i += 256) bt[i] = btab[i * H_ + h];

  // Q fragment (A layout) direct from global bf16
  AF qf;
  {
    int qrow = qbase + ln;
    const __bf16* qp = q + ((size_t)(b * S_ + qrow)) * EMB_ + h * 32;
    qf.h[0] = *reinterpret_cast<const v8bf*>(qp + half * 8);
    qf.h[1] = *reinterpret_cast<const v8bf*>(qp + 16 + half * 8);
  }

  float rm[8], rs[8];
  v8f acc1[2], acc2[2];
#pragma unroll
  for (int e = 0; e < 8; ++e) {
    rm[e] = -1e30f; rs[e] = 0.0f;
    acc1[0][e] = 0.0f; acc1[1][e] = 0.0f;
    acc2[0][e] = 0.0f; acc2[1][e] = 0.0f;
  }
  __syncthreads();   // bt ready

  for (int kc = 0; kc < 8; ++kc) {
    const int kb = kc * 64;
    // load 64 key rows + transposed V chunk
    {
      int r = tid >> 2, c = (tid & 3) << 3;
      size_t base = ((size_t)(b * S_ + kb + r)) * EMB_ + h * 32 + c;
#if USE_ASYNC_COPY
      ASYNC_B128(&k[base], &Ks[r][c]);
#else
      *reinterpret_cast<v8bf*>(&Ks[r][c]) = *reinterpret_cast<const v8bf*>(k + base);
#endif
      v8bf vv = *reinterpret_cast<const v8bf*>(v + base);
#pragma unroll
      for (int j = 0; j < 8; ++j) VsT[c + j][r] = vv[j];
#if USE_ASYNC_COPY
      ASYNC_WAIT();
#endif
    }
    __syncthreads();

    // scores chunk: 16 x 64, then apply softmax scale EMB^-0.5 = 1/16
    v8f sc[4];
#pragma unroll
    for (int nt = 0; nt < 4; ++nt) {
      AF bf;
      const __bf16* p = &Ks[nt * 16 + ln][0];
      bf.h[0] = *reinterpret_cast<const v8bf*>(p + half * 8);
      bf.h[1] = *reinterpret_cast<const v8bf*>(p + 16 + half * 8);
      v8f z;
#pragma unroll
      for (int e = 0; e < 8; ++e) z[e] = 0.0f;
      sc[nt] = __builtin_amdgcn_wmma_f32_16x16x32_bf16(
          false, qf.v, false, bf.v, (short)0, z, false, false);
#pragma unroll
      for (int e = 0; e < 8; ++e) sc[nt][e] *= 0.0625f;
    }

    // V B-fragments (consumed by both P@V and RB@V)
    AF vf[2][2];
#pragma unroll
    for (int nt2 = 0; nt2 < 2; ++nt2)
#pragma unroll
      for (int kc2 = 0; kc2 < 2; ++kc2) {
        const __bf16* p = &VsT[nt2 * 16 + ln][0];
        vf[nt2][kc2].h[0] = *reinterpret_cast<const v8bf*>(p + kc2 * 32 + half * 8);
        vf[nt2][kc2].h[1] = *reinterpret_cast<const v8bf*>(p + kc2 * 32 + 16 + half * 8);
      }

    // ---- batched online softmax: all 8 rows per shuffle step --------------
    float lm[8];
#pragma unroll
    for (int e = 0; e < 8; ++e)
      lm[e] = fmaxf(fmaxf(sc[0][e], sc[1][e]), fmaxf(sc[2][e], sc[3][e]));
#pragma unroll
    for (int mask = 1; mask < 16; mask <<= 1) {
      float tmp[8];
#pragma unroll
      for (int e = 0; e < 8; ++e) tmp[e] = __shfl_xor(lm[e], mask, 16);
#pragma unroll
      for (int e = 0; e < 8; ++e) lm[e] = fmaxf(lm[e], tmp[e]);
    }
    float ls[8];
#pragma unroll
    for (int e = 0; e < 8; ++e) {
      float mnew = fmaxf(rm[e], lm[e]);
      float f = __expf(rm[e] - mnew);
      rs[e] *= f; acc1[0][e] *= f; acc1[1][e] *= f;
      rm[e] = mnew;
      float s = 0.0f;
#pragma unroll
      for (int nt = 0; nt < 4; ++nt) {
        float pp = __expf(sc[nt][e] - mnew);
        sc[nt][e] = pp; s += pp;
      }
      ls[e] = s;
    }
#pragma unroll
    for (int mask = 1; mask < 16; mask <<= 1) {
      float tmp[8];
#pragma unroll
      for (int e = 0; e < 8; ++e) tmp[e] = __shfl_xor(ls[e], mask, 16);
#pragma unroll
      for (int e = 0; e < 8; ++e) ls[e] += tmp[e];
    }
#pragma unroll
    for (int e = 0; e < 8; ++e) rs[e] += ls[e];

    // stage P (C layout -> LDS) so it can be re-read in A layout
#pragma unroll
    for (int nt = 0; nt < 4; ++nt)
#pragma unroll
      for (int e = 0; e < 8; ++e)
        Ps[wid][e + 8 * half][nt * 16 + ln] = f2bf(sc[nt][e]);
    __syncthreads();   // also guards Ks/VsT reuse next chunk

    AF pf[2];
#pragma unroll
    for (int kc2 = 0; kc2 < 2; ++kc2) {
      const __bf16* p = &Ps[wid][ln][kc2 * 32];
      pf[kc2].h[0] = *reinterpret_cast<const v8bf*>(p + half * 8);
      pf[kc2].h[1] = *reinterpret_cast<const v8bf*>(p + 16 + half * 8);
    }
    // relative-bias A-fragments gathered from bt
    AF rbf[2];
#pragma unroll
    for (int kc2 = 0; kc2 < 2; ++kc2)
#pragma unroll
      for (int e = 0; e < 16; ++e) {
        int kk = kc2 * 32 + (e < 8 ? half * 8 + e : 16 + half * 8 + (e - 8));
        int d = (qbase + ln) - (kb + kk) + (S_ - 1);
        rbf[kc2].v[e] = f2bf(bt[d]);
      }

#pragma unroll
    for (int nt2 = 0; nt2 < 2; ++nt2)
#pragma unroll
      for (int kc2 = 0; kc2 < 2; ++kc2) {
        acc1[nt2] = __builtin_amdgcn_wmma_f32_16x16x32_bf16(
            false, pf[kc2].v, false, vf[nt2][kc2].v, (short)0, acc1[nt2], false, false);
        acc2[nt2] = __builtin_amdgcn_wmma_f32_16x16x32_bf16(
            false, rbf[kc2].v, false, vf[nt2][kc2].v, (short)0, acc2[nt2], false, false);
      }
  }

  // fused residual write: x += attn_out ; refresh bf16 shadow
#pragma unroll
  for (int nt2 = 0; nt2 < 2; ++nt2)
#pragma unroll
    for (int e = 0; e < 8; ++e) {
      int qrow = qbase + e + 8 * half;
      int col  = h * 32 + nt2 * 16 + ln;
      size_t idx = ((size_t)(b * S_ + qrow)) * EMB_ + col;
      float o = x[idx] + acc1[nt2][e] / rs[e] + acc2[nt2][e];
      x[idx]  = o;
      xb[idx] = f2bf(o);
    }
}

// ---------------------------------------------------------------------------
__global__ __launch_bounds__(256) void addln_kernel(
    float* __restrict__ x, __bf16* __restrict__ xb, const float* __restrict__ ff,
    const float* __restrict__ g, const float* __restrict__ beta)
{
  __shared__ float red[16];
  int row = blockIdx.x, e = threadIdx.x;
  size_t idx = (size_t)row * EMB_ + e;
  float val = x[idx] + ff[idx];
  float s = val, s2 = val * val;
#pragma unroll
  for (int m = 1; m < 32; m <<= 1) {
    s  += __shfl_xor(s,  m, 32);
    s2 += __shfl_xor(s2, m, 32);
  }
  int wid = e >> 5, lane = e & 31;
  if (lane == 0) { red[wid] = s; red[8 + wid] = s2; }
  __syncthreads();
  s = 0.0f; s2 = 0.0f;
#pragma unroll
  for (int w = 0; w < 8; ++w) { s += red[w]; s2 += red[8 + w]; }
  float mu  = s * (1.0f / EMB_);
  float var = s2 * (1.0f / EMB_) - mu * mu;
  float o = (val - mu) * rsqrtf(var + 1e-5f) * g[e] + beta[e];
  x[idx]  = o;
  xb[idx] = f2bf(o);
}

// wave-per-row dot product: out[row] = h[row,:] . Wout + bout
__global__ __launch_bounds__(256) void head_kernel(
    const float* __restrict__ hk, const float* __restrict__ Wout,
    const float* __restrict__ bout, float* __restrict__ out)
{
  int lane = threadIdx.x & 31, wid = threadIdx.x >> 5;
  int row = blockIdx.x * 8 + wid;
  const float* hp = hk + (size_t)row * EMB_;
  float acc = 0.0f;
#pragma unroll
  for (int j = 0; j < 8; ++j) acc += hp[lane + j * 32] * Wout[lane + j * 32];
#pragma unroll
  for (int m = 1; m < 32; m <<= 1) acc += __shfl_xor(acc, m, 32);
  if (lane == 0) out[row] = acc + bout[0];
}

// ---------------------------------------------------------------------------
extern "C" void kernel_launch(void* const* d_in, const int* in_sizes, int n_in,
                              void* d_out, int out_size, void* d_ws, size_t ws_size,
                              hipStream_t stream) {
  (void)in_sizes; (void)n_in; (void)out_size; (void)ws_size;
  const float* src   = (const float*)d_in[0];
  const float* Win   = (const float*)d_in[4];
  const float* b_in  = (const float*)d_in[5];
  const float* Wq    = (const float*)d_in[6];
  const float* Wk    = (const float*)d_in[7];
  const float* Wv    = (const float*)d_in[8];
  const float* btab  = (const float*)d_in[9];
  const float* W1    = (const float*)d_in[10];
  const float* b1    = (const float*)d_in[11];
  const float* W2    = (const float*)d_in[12];
  const float* b2    = (const float*)d_in[13];
  const float* W3    = (const float*)d_in[14];
  const float* b3    = (const float*)d_in[15];
  const float* g2    = (const float*)d_in[16];
  const float* beta2 = (const float*)d_in[17];
  const float* fW1   = (const float*)d_in[18];
  const float* fb1   = (const float*)d_in[19];
  const float* fW2   = (const float*)d_in[20];
  const float* fb2   = (const float*)d_in[21];
  const float* fW3   = (const float*)d_in[22];
  const float* fb3   = (const float*)d_in[23];
  const float* Wout  = (const float*)d_in[24];
  const float* bout  = (const float*)d_in[25];
  float* out = (float*)d_out;

  // ---- workspace carve-up -------------------------------------------------
  const size_t n_xe = (size_t)NROW * EMB_;     // 4.19M
  const size_t n_xf = (size_t)NROW * DFF_;     // 16.8M
  char* wsb = (char*)d_ws;
  float*  x    = (float*)wsb;                wsb += n_xe * 4;
  float*  ff   = (float*)wsb;                wsb += n_xe * 4;
  __bf16* xb   = (__bf16*)wsb;               wsb += n_xe * 2;
  __bf16* qb   = (__bf16*)wsb;               wsb += n_xe * 2;
  __bf16* kb   = (__bf16*)wsb;               wsb += n_xe * 2;
  __bf16* vb   = (__bf16*)wsb;               wsb += n_xe * 2;
  __bf16* h1b  = (__bf16*)wsb;               wsb += n_xf * 2;
  __bf16* h2b  = (__bf16*)wsb;               wsb += n_xf * 2;
  const size_t nEE = (size_t)L_ * EMB_ * EMB_;   // 262144
  const size_t nEF = (size_t)L_ * EMB_ * DFF_;   // 1048576
  const size_t nFF = (size_t)L_ * DFF_ * DFF_;   // 4194304
  __bf16* WqT  = (__bf16*)wsb;               wsb += nEE * 2;
  __bf16* WkT  = (__bf16*)wsb;               wsb += nEE * 2;
  __bf16* WvT  = (__bf16*)wsb;               wsb += nEE * 2;
  __bf16* W1T  = (__bf16*)wsb;               wsb += nEF * 2;
  __bf16* W2T  = (__bf16*)wsb;               wsb += nFF * 2;
  __bf16* W3T  = (__bf16*)wsb;               wsb += nEF * 2;
  __bf16* fW1T = (__bf16*)wsb;               wsb += ((size_t)EMB_ * DFF_) * 2;
  __bf16* fW2T = (__bf16*)wsb;               wsb += ((size_t)DFF_ * DFF_) * 2;
  __bf16* fW3T = (__bf16*)wsb;               wsb += ((size_t)DFF_ * EMB_) * 2;

  dim3 blk(256);

  // ---- weight conversion + transpose (bf16 [N][K]) ------------------------
  cvt_t_kernel<<<dim3((unsigned)(nEE / 256)), blk, 0, stream>>>(Wq, WqT, EMB_, EMB_);
  cvt_t_kernel<<<dim3((unsigned)(nEE / 256)), blk, 0, stream>>>(Wk, WkT, EMB_, EMB_);
  cvt_t_kernel<<<dim3((unsigned)(nEE / 256)), blk, 0, stream>>>(Wv, WvT, EMB_, EMB_);
  cvt_t_kernel<<<dim3((unsigned)(nEF / 256)), blk, 0, stream>>>(W1, W1T, EMB_, DFF_);
  cvt_t_kernel<<<dim3((unsigned)(nFF / 256)), blk, 0, stream>>>(W2, W2T, DFF_, DFF_);
  cvt_t_kernel<<<dim3((unsigned)(nEF / 256)), blk, 0, stream>>>(W3, W3T, DFF_, EMB_);
  cvt_t_kernel<<<dim3((unsigned)((size_t)EMB_ * DFF_ / 256)), blk, 0, stream>>>(fW1, fW1T, EMB_, DFF_);
  cvt_t_kernel<<<dim3((unsigned)((size_t)DFF_ * DFF_ / 256)), blk, 0, stream>>>(fW2, fW2T, DFF_, DFF_);
  cvt_t_kernel<<<dim3((unsigned)((size_t)DFF_ * EMB_ / 256)), blk, 0, stream>>>(fW3, fW3T, DFF_, EMB_);

  // ---- forward ------------------------------------------------------------
  embed_kernel<<<dim3((unsigned)(n_xe / 256)), blk, 0, stream>>>(src, Win, b_in, x, xb);

  for (int l = 0; l < L_; ++l) {
    dim3 g_e(EMB_ / 128, NROW / 256);
    dim3 g_f(DFF_ / 128, NROW / 256);
    gemm_wmma<false, false, false, true><<<g_e, blk, 0, stream>>>(
        xb, WqT + (size_t)l * EMB_ * EMB_, nullptr, nullptr, qb, NROW, EMB_, EMB_);
    gemm_wmma<false, false, false, true><<<g_e, blk, 0, stream>>>(
        xb, WkT + (size_t)l * EMB_ * EMB_, nullptr, nullptr, kb, NROW, EMB_, EMB_);
    gemm_wmma<false, false, false, true><<<g_e, blk, 0, stream>>>(
        xb, WvT + (size_t)l * EMB_ * EMB_, nullptr, nullptr, vb, NROW, EMB_, EMB_);
    attn_kernel<<<dim3(B_ * H_ * 4), blk, 0, stream>>>(
        qb, kb, vb, btab + (size_t)l * (2 * S_ - 1) * H_, x, xb);
    gemm_wmma<true,  true,  false, true><<<g_f, blk, 0, stream>>>(
        xb,  W1T + (size_t)l * EMB_ * DFF_, b1 + (size_t)l * DFF_, nullptr, h1b, NROW, DFF_, EMB_);
    gemm_wmma<true,  true,  false, true><<<g_f, blk, 0, stream>>>(
        h1b, W2T + (size_t)l * DFF_ * DFF_, b2 + (size_t)l * DFF_, nullptr, h2b, NROW, DFF_, DFF_);
    gemm_wmma<false, true,  true,  false><<<g_e, blk, 0, stream>>>(
        h2b, W3T + (size_t)l * DFF_ * EMB_, b3 + (size_t)l * EMB_, ff, nullptr, NROW, EMB_, DFF_);
    addln_kernel<<<dim3(NROW), blk, 0, stream>>>(
        x, xb, ff, g2 + (size_t)l * EMB_, beta2 + (size_t)l * EMB_);
  }

  gemm_wmma<true,  true,  false, true><<<dim3(DFF_ / 128, NROW / 256), blk, 0, stream>>>(
      xb,  fW1T, fb1, nullptr, h1b, NROW, DFF_, EMB_);
  gemm_wmma<true,  true,  false, true><<<dim3(DFF_ / 128, NROW / 256), blk, 0, stream>>>(
      h1b, fW2T, fb2, nullptr, h2b, NROW, DFF_, DFF_);
  gemm_wmma<false, true,  true,  false><<<dim3(EMB_ / 128, NROW / 256), blk, 0, stream>>>(
      h2b, fW3T, fb3, ff, nullptr, NROW, EMB_, DFF_);
  head_kernel<<<dim3(NROW / 8), blk, 0, stream>>>(ff, Wout, bout, out);
}
